// model_bertcrf_86629490360927
// MI455X (gfx1250) — matrix operand order repaired and checked by reading, not compiled, not run on hardware
//
#include <hip/hip_runtime.h>
#include <hip/hip_bf16.h>

typedef __attribute__((ext_vector_type(16))) __bf16 v16bf;
typedef __attribute__((ext_vector_type(8)))  float  v8f;

#define T_LEN 4096
#define H_DIM 768
#define L_LAB 27
#define START_TAG 25
#define STOP_TAG 26
#define LPAD 32
#define KPAD (H_DIM + 8)   // pad K stride in LDS to break bank alignment

// ---------------------------------------------------------------------------
// Kernel 1: logits[4096][32] = hidden[0] @ W (bf16 WMMA, f32 accumulate) + b
//   grid = 32 blocks x 256 threads (8 waves); wave w owns rows [blk*128+w*16, +16)
// ---------------------------------------------------------------------------
__global__ void __launch_bounds__(256)
bertcrf_gemm_logits(const float* __restrict__ hidden,
                    const float* __restrict__ W,     // [768][27] row-major
                    const float* __restrict__ bias,  // [27]
                    float* __restrict__ logits)      // [4096][32] padded
{
    // Stage W into LDS transposed [N][K] as bf16, zero-padded to N=32.
    __shared__ __bf16 sW[LPAD][KPAD];
    const int tid = threadIdx.x;
    for (int idx = tid; idx < H_DIM * LPAD; idx += blockDim.x) {
        const int k = idx >> 5;       // 0..767
        const int n = idx & 31;       // 0..31
        const float v = (n < L_LAB) ? W[k * L_LAB + n] : 0.0f;
        sW[n][k] = (__bf16)v;
    }
    __syncthreads();

    const int wave = tid >> 5;
    const int lane = tid & 31;
    const int half = lane >> 4;       // 0: lanes 0-15, 1: lanes 16-31
    const int mr   = lane & 15;
    const int m0   = blockIdx.x * 128 + wave * 16;

    const float* __restrict__ Arow = hidden + (size_t)(m0 + mr) * H_DIM;

    v8f c0 = {}; // N tile 0 (cols 0-15)
    v8f c1 = {}; // N tile 1 (cols 16-31)

    for (int kb = 0; kb < H_DIM; kb += 32) {
        // A fragment: contiguous K order (matches B) ->
        //   lanes 0-15 : K = kb+0..15  (row M = lane)
        //   lanes 16-31: K = kb+16..31 (row M = lane-16)
        const int k0 = kb + half * 16;
        const int k1 = kb + half * 16 + 8;
        v16bf a;
#pragma unroll
        for (int e = 0; e < 8; ++e) a[e]     = (__bf16)Arow[k0 + e];
#pragma unroll
        for (int e = 0; e < 8; ++e) a[8 + e] = (__bf16)Arow[k1 + e];

        // B fragment: lanes 0-15 hold K=kb..kb+15 at N=lane;
        //             lanes 16-31 hold K=kb+16..kb+31 at N=lane-16.
        const int kB = kb + half * 16;
        v16bf b0, b1;
#pragma unroll
        for (int e = 0; e < 16; ++e) b0[e] = sW[mr][kB + e];
#pragma unroll
        for (int e = 0; e < 16; ++e) b1[e] = sW[16 + mr][kB + e];

        c0 = __builtin_amdgcn_wmma_f32_16x16x32_bf16(false, a, false, b0,
                                                     (short)0, c0, false, false);
        c1 = __builtin_amdgcn_wmma_f32_16x16x32_bf16(false, a, false, b1,
                                                     (short)0, c1, false, false);
    }

    // C/D layout: VGPR r -> row m0 + r + half*8, col = (ntile*16 + mr)
#pragma unroll
    for (int r = 0; r < 8; ++r) {
        const int row = m0 + r + half * 8;
        const int n0  = mr;            // always < 27
        const int n1  = 16 + mr;       // may be >= 27 (padding)
        logits[row * LPAD + n0] = c0[r] + bias[n0];
        logits[row * LPAD + n1] = c1[r] + ((n1 < L_LAB) ? bias[n1] : 0.0f);
    }
}

// ---------------------------------------------------------------------------
// Kernel 2: Viterbi decode on logits[0..4095][0..26]. One wave32; lane j = tag.
//   out[0] = path_score, out[1..4096] = best_path (as float)
// ---------------------------------------------------------------------------
__global__ void __launch_bounds__(32)
bertcrf_viterbi(const float* __restrict__ logits,  // [4096][32]
                const float* __restrict__ trans,   // [27][27]
                unsigned char* __restrict__ bps,   // [4096][32] (row 0 unused)
                float* __restrict__ out)           // [4097]
{
    const int j = threadIdx.x;            // 0..31; tags are j < 27
    const bool active = (j < L_LAB);

    __shared__ float sAlpha[LPAD];

    // trans[:, j] resident in registers for this lane
    float tj[L_LAB];
#pragma unroll
    for (int f = 0; f < L_LAB; ++f)
        tj[f] = active ? trans[f * L_LAB + j] : -3.0e38f;

    // alpha0 = logits[0] + trans[START]
    float a = active ? (logits[j] + trans[START_TAG * L_LAB + j]) : -3.0e38f;
    sAlpha[j] = a;
    __syncthreads();

    for (int t = 1; t < T_LEN; ++t) {
        float m = -3.4e38f;
        int bp = 0;
#pragma unroll
        for (int f = 0; f < L_LAB; ++f) {
            const float v = sAlpha[f] + tj[f];
            if (v > m) { m = v; bp = f; }
        }
        const float anew = active ? (m + logits[t * LPAD + j]) : -3.0e38f;
        bps[t * LPAD + j] = (unsigned char)bp;
        __syncthreads();
        sAlpha[j] = anew;
        __syncthreads();
        a = anew;
    }

    // final = alpha + trans[:, STOP]; argmax with first-index tie-break
    float fin = active ? (a + trans[j * L_LAB + STOP_TAG]) : -3.4e38f;
    int idx = j;
#pragma unroll
    for (int off = 16; off > 0; off >>= 1) {
        const float om = __shfl_xor(fin, off, 32);
        const int   oi = __shfl_xor(idx, off, 32);
        if (om > fin || (om == fin && oi < idx)) { fin = om; idx = oi; }
    }

    if (j == 0) {
        out[0] = fin;
        int cur = idx;
        out[1 + (T_LEN - 1)] = (float)cur;
        for (int t = T_LEN - 1; t >= 1; --t) {
            cur = (int)bps[t * LPAD + cur];
            out[1 + (t - 1)] = (float)cur;
        }
    }
}

// ---------------------------------------------------------------------------
extern "C" void kernel_launch(void* const* d_in, const int* in_sizes, int n_in,
                              void* d_out, int out_size, void* d_ws, size_t ws_size,
                              hipStream_t stream) {
    (void)in_sizes; (void)n_in; (void)out_size; (void)ws_size;

    const float* hidden = (const float*)d_in[0];  // [8,4096,768]; batch 0 only
    // d_in[1] = attention_mask (all ones; unused, matches reference semantics)
    const float* W      = (const float*)d_in[2];  // [768,27]
    const float* bias   = (const float*)d_in[3];  // [27]
    const float* trans  = (const float*)d_in[4];  // [27,27]

    float* logits      = (float*)d_ws;                                    // 4096*32 f32
    unsigned char* bps = (unsigned char*)d_ws + (size_t)T_LEN * LPAD * 4; // 4096*32 u8
    float* out         = (float*)d_out;                                   // [4097]

    bertcrf_gemm_logits<<<T_LEN / 128, 256, 0, stream>>>(hidden, W, bias, logits);
    bertcrf_viterbi<<<1, 32, 0, stream>>>(logits, trans, bps, out);
}
